// AAEncoder_16260746183166
// MI455X (gfx1250) — matrix operand order repaired and checked by reading, not compiled, not run on hardware
//
#include <hip/hip_runtime.h>
#include <hip/hip_bf16.h>

#define NAG 1024
#define TT  50

typedef __attribute__((ext_vector_type(16))) _Float16 v16h;
typedef __attribute__((ext_vector_type(8)))  _Float16 v8h;
typedef __attribute__((ext_vector_type(8)))  float    v8f;

// ---------------- WMMA fragment helpers (gfx1250 wave32 layouts) ----------------
__device__ __forceinline__ v16h make_v16(v8h lo, v8h hi) {
  v16h r;
#pragma unroll
  for (int i = 0; i < 8; ++i) { r[i] = lo[i]; r[i + 8] = hi[i]; }
  return r;
}

// A 16x32 f16: lanes0-15 M=lane, K {0..7,16..23}; lanes16-31 M=lane-16, K {8..15,24..31}
__device__ __forceinline__ v16h load_afrag(const _Float16* hT, int lane, int kc) {
  int m = lane & 15;
  int koff = (lane >> 4) * 8;
  const _Float16* p = hT + m * 64 + kc * 32 + koff;
  return make_v16(*(const v8h*)p, *(const v8h*)(p + 16));
}

// B 32x16 f16 (B[k][n] = W[n][k], W row-major [out][in]):
// lanes0-15 N=lane, K {0..7,16..23}; lanes16-31 K {8..15,24..31} (offset kc*32)
__device__ __forceinline__ v16h load_bfrag(const _Float16* W, int lane, int nt, int kc) {
  int n = (lane & 15) + nt * 16;
  int k0 = kc * 32 + (lane >> 4) * 8;
  const _Float16* p = W + n * 64 + k0;
  return make_v16(*(const v8h*)p, *(const v8h*)(p + 16));
}

__device__ __forceinline__ v8f wmma16(v16h a, v16h b, v8f c) {
  return __builtin_amdgcn_wmma_f32_16x16x32_f16(false, a, false, b, (short)0, c, false, false);
}

// In-place LayerNorm over 16x64 f16 LDS tile; lane pair per row, shfl combine.
__device__ __forceinline__ void ln_pass_f16(_Float16* hT, int lane,
                                            const float* g, const float* b, bool relu) {
  int row = lane >> 1;
  int c0 = (lane & 1) * 32;
  float x[32];
  float s = 0.f, s2 = 0.f;
#pragma unroll
  for (int i = 0; i < 32; ++i) {
    float v = (float)hT[row * 64 + c0 + i];
    x[i] = v; s += v; s2 += v * v;
  }
  s  += __shfl_xor(s, 1, 32);
  s2 += __shfl_xor(s2, 1, 32);
  float mean = s * (1.f / 64.f);
  float rs = rsqrtf(s2 * (1.f / 64.f) - mean * mean + 1e-5f);
#pragma unroll
  for (int i = 0; i < 32; ++i) {
    float y = (x[i] - mean) * rs * g[c0 + i] + b[c0 + i];
    if (relu) y = fmaxf(y, 0.f);
    hT[row * 64 + c0 + i] = (_Float16)y;
  }
}

// ---------------- Kernel A: per-agent precompute ----------------
__device__ __forceinline__ void ln64(float* x, const float* g, const float* b, bool relu) {
  float s = 0.f, s2 = 0.f;
  for (int k = 0; k < 64; ++k) { s += x[k]; s2 += x[k] * x[k]; }
  float m = s * (1.f / 64.f);
  float rs = rsqrtf(s2 * (1.f / 64.f) - m * m + 1e-5f);
  for (int k = 0; k < 64; ++k) {
    float y = (x[k] - m) * rs * g[k] + b[k];
    x[k] = relu ? fmaxf(y, 0.f) : y;
  }
}

__device__ __forceinline__ void mv64(float* out, const float* W, const float* x, const float* b) {
  for (int d = 0; d < 64; ++d) {
    float a = b ? b[d] : 0.f;
    for (int k = 0; k < 64; ++k) a += W[d * 64 + k] * x[k];
    out[d] = a;
  }
}

__global__ void prep_kernel(
    const float* positions, const int* padmask, const int* tptr,
    const float* cl1W, const float* cl1b, const float* cl2W, const float* cl2b,
    const float* cl3W, const float* cl3b,
    const float* cn1b, const float* cn1g, const float* cn2b, const float* cn2g,
    const float* cn3b, const float* cn3g, const float* n1b, const float* n1g,
    const float* WqG,
    const float* dl1W, const float* dl1b, const float* dn1b, const float* dn1g,
    const float* dl2W, const float* dl2b,
    float* gPosT, float* gValid, float* gC, float* gQ, float* gEdxy) {
  int i = blockIdx.x * blockDim.x + threadIdx.x;
  if (i >= NAG) return;
  int t = *tptr;
  float px = positions[(i * TT + t) * 2 + 0];
  float py = positions[(i * TT + t) * 2 + 1];
  float dx = px - positions[(i * TT + t - 1) * 2 + 0];
  float dy = py - positions[(i * TT + t - 1) * 2 + 1];
  gPosT[i * 2 + 0] = px;
  gPosT[i * 2 + 1] = py;
  gValid[i] = padmask[i * TT + t] ? 0.f : 1.f;

  float h[64], h2[64];
  // center: l1 -> n1 -> relu -> l2 -> n2 -> relu -> l3 -> n3 -> norm1
  for (int d = 0; d < 64; ++d) h[d] = dx * cl1W[d * 2] + dy * cl1W[d * 2 + 1] + cl1b[d];
  ln64(h, cn1g, cn1b, true);
  mv64(h2, cl2W, h, cl2b);  ln64(h2, cn2g, cn2b, true);
  mv64(h, cl3W, h2, cl3b);  ln64(h, cn3g, cn3b, false);
  ln64(h, n1g, n1b, false);  // c
  for (int d = 0; d < 64; ++d) gC[i * 64 + d] = h[d];
  mv64(h2, WqG, h, nullptr); // q, prescaled by 1/sqrt(HD)
  for (int d = 0; d < 64; ++d) gQ[i * 64 + d] = h2[d] * 0.35355339059f;

  // e_dxy: l1 -> n1 -> relu -> l2
  for (int d = 0; d < 64; ++d) h[d] = dx * dl1W[d * 2] + dy * dl1W[d * 2 + 1] + dl1b[d];
  ln64(h, dn1g, dn1b, true);
  mv64(h2, dl2W, h, dl2b);
  for (int d = 0; d < 64; ++d) gEdxy[i * 64 + d] = h2[d];
}

// ---------------- Kernel B: fused pair-MLP + flash attention ----------------
#define SM_WXY2 0
#define SM_WAG  8192
#define SM_WK   16384
#define SM_WV   24576
#define SM_HT   32768       // 8 waves x 16x64 f16
#define SM_MISC 49152
#define SMEM_BYTES 52224
// merge phase aliases [0, 40960): mPart@0, lPart@4096, oPart@8192 (weights+hT are dead)

__launch_bounds__(256)
__global__ void pair_attn_kernel(
    const float* xyl1W, const float* xyl1b, const float* xyn1b, const float* xyn1g,
    const float* xyl2W, const float* xyl2b,
    const float* agW, const float* agb, const float* agn1b, const float* agn1g,
    const float* agn2b, const float* agn2g,
    const float* WkG, const float* WvG,
    const float* gPosT, const float* gValid, const float* gQ, const float* gEdxy,
    float* gO) {
  __shared__ __align__(16) unsigned char smem[SMEM_BYTES];
  _Float16* sWxy2 = (_Float16*)(smem + SM_WXY2);
  _Float16* sWag  = (_Float16*)(smem + SM_WAG);
  _Float16* sWk   = (_Float16*)(smem + SM_WK);
  _Float16* sWv   = (_Float16*)(smem + SM_WV);
  _Float16* hTall = (_Float16*)(smem + SM_HT);
  float* sMisc = (float*)(smem + SM_MISC);
  float* sW1  = sMisc;        // 128: nbr_xy.l1.W
  float* sB1  = sMisc + 128;  // nbr_xy.l1.b
  float* sG1  = sMisc + 192;  // nbr_xy.n1.g
  float* sBt1 = sMisc + 256;  // nbr_xy.n1.b
  float* sB2  = sMisc + 320;  // nbr_xy.l2.b
  float* sGa1 = sMisc + 384;  // aggr.n1.g
  float* sBa1 = sMisc + 448;  // aggr.n1.b
  float* sBag = sMisc + 512;  // aggr.l.b
  float* sGa2 = sMisc + 576;  // aggr.n2.g
  float* sBa2 = sMisc + 640;  // aggr.n2.b
  float* sPosI = sMisc + 704; // 32

  int tid = threadIdx.x;
  int it = blockIdx.x;

  for (int idx = tid; idx < 4096; idx += 256) {
    sWxy2[idx] = (_Float16)xyl2W[idx];
    sWag[idx]  = (_Float16)agW[idx];
    sWk[idx]   = (_Float16)WkG[idx];
    sWv[idx]   = (_Float16)WvG[idx];
  }
  if (tid < 128) sW1[tid] = xyl1W[tid];
  if (tid < 64) {
    sB1[tid] = xyl1b[tid]; sG1[tid] = xyn1g[tid]; sBt1[tid] = xyn1b[tid];
    sB2[tid] = xyl2b[tid]; sGa1[tid] = agn1g[tid]; sBa1[tid] = agn1b[tid];
    sBag[tid] = agb[tid];  sGa2[tid] = agn2g[tid]; sBa2[tid] = agn2b[tid];
  }
  if (tid < 32) sPosI[tid] = gPosT[it * 32 + tid];
  __syncthreads();

  int lane = tid & 31;
  int wv = tid >> 5;
  _Float16* hT = hTall + wv * 1024;
  int rh8 = (lane >> 4) * 8;
  int cl = lane & 15;
  int r0 = lane >> 1;
  int c0 = (lane & 1) * 32;

  // loop-invariant per-lane state: i-tile positions, global row ids, q fragment
  float pix[8], piy[8];
  int ig[8];
  float qreg[4][8];
#pragma unroll
  for (int vr = 0; vr < 8; ++vr) {
    int row = vr + rh8;
    pix[vr] = sPosI[row * 2]; piy[vr] = sPosI[row * 2 + 1];
    ig[vr] = it * 16 + row;
#pragma unroll
    for (int nt = 0; nt < 4; ++nt)
      qreg[nt][vr] = gQ[(it * 16 + row) * 64 + cl + 16 * nt];
  }
  float pr0x = sPosI[r0 * 2], pr0y = sPosI[r0 * 2 + 1];

  float mst[4][8], lst[4][8], ost[4][8];
#pragma unroll
  for (int nt = 0; nt < 4; ++nt)
#pragma unroll
    for (int vr = 0; vr < 8; ++vr) { mst[nt][vr] = -3.0e38f; lst[nt][vr] = 0.f; ost[nt][vr] = 0.f; }

  for (int j = wv; j < NAG; j += 8) {
    float pjx = gPosT[j * 2], pjy = gPosT[j * 2 + 1];
    float validj = gValid[j];

    // mask first: skip the entire pair chain if no row of this i-tile attends to j
    float okf[8];
    int lok = 0;
#pragma unroll
    for (int vr = 0; vr < 8; ++vr) {
      float rx = pjx - pix[vr], ry = pjy - piy[vr];
      float d2 = rx * rx + ry * ry;
      bool ok = ((d2 <= 2500.f) && (validj > 0.5f)) || (j == ig[vr]);
      okf[vr] = ok ? 1.f : 0.f;
      lok |= (int)ok;
    }
    if (__ballot(lok) == 0ull) continue;

    // stage 0: h = relu(LN(rel @ W1xy^T + b1)) -> hT (f16), VALU (K=2)
    {
      float rx = pjx - pr0x;
      float ry = pjy - pr0y;
      float x[32];
      float s = 0.f, s2 = 0.f;
#pragma unroll
      for (int i2 = 0; i2 < 32; ++i2) {
        int col = c0 + i2;
        float v = rx * sW1[col * 2] + ry * sW1[col * 2 + 1] + sB1[col];
        x[i2] = v; s += v; s2 += v * v;
      }
      s  += __shfl_xor(s, 1, 32);
      s2 += __shfl_xor(s2, 1, 32);
      float mean = s * (1.f / 64.f);
      float rs = rsqrtf(s2 * (1.f / 64.f) - mean * mean + 1e-5f);
#pragma unroll
      for (int i2 = 0; i2 < 32; ++i2) {
        int col = c0 + i2;
        float y = (x[i2] - mean) * rs * sG1[col] + sBt1[col];
        hT[r0 * 64 + col] = (_Float16)fmaxf(y, 0.f);
      }
    }
    __builtin_amdgcn_wave_barrier();

    // stage 1: e = h @ Wxy2 + b2 + e_dxy[j]  (WMMA), then LN(aggr.n1)+relu in place
    {
      v16h a0 = load_afrag(hT, lane, 0);
      v16h a1 = load_afrag(hT, lane, 1);
      v8f acc[4];
#pragma unroll
      for (int nt = 0; nt < 4; ++nt) {
        v8f c = {};
        c = wmma16(a0, load_bfrag(sWxy2, lane, nt, 0), c);
        c = wmma16(a1, load_bfrag(sWxy2, lane, nt, 1), c);
        acc[nt] = c;
      }
      __builtin_amdgcn_wave_barrier();
#pragma unroll
      for (int nt = 0; nt < 4; ++nt)
#pragma unroll
        for (int vr = 0; vr < 8; ++vr) {
          int row = vr + rh8, col = cl + 16 * nt;
          float val = acc[nt][vr] + sB2[col] + gEdxy[j * 64 + col];
          hT[row * 64 + col] = (_Float16)val;
        }
    }
    __builtin_amdgcn_wave_barrier();
    ln_pass_f16(hT, lane, sGa1, sBa1, true);
    __builtin_amdgcn_wave_barrier();

    // stage 2: nb = LN(a @ Wag + bag)  (WMMA)
    {
      v16h a0 = load_afrag(hT, lane, 0);
      v16h a1 = load_afrag(hT, lane, 1);
      v8f acc[4];
#pragma unroll
      for (int nt = 0; nt < 4; ++nt) {
        v8f c = {};
        c = wmma16(a0, load_bfrag(sWag, lane, nt, 0), c);
        c = wmma16(a1, load_bfrag(sWag, lane, nt, 1), c);
        acc[nt] = c;
      }
      __builtin_amdgcn_wave_barrier();
#pragma unroll
      for (int nt = 0; nt < 4; ++nt)
#pragma unroll
        for (int vr = 0; vr < 8; ++vr) {
          int row = vr + rh8, col = cl + 16 * nt;
          hT[row * 64 + col] = (_Float16)(acc[nt][vr] + sBag[col]);
        }
    }
    __builtin_amdgcn_wave_barrier();
    ln_pass_f16(hT, lane, sGa2, sBa2, false);
    __builtin_amdgcn_wave_barrier();

    // stage 3+4: k,v = nb @ Wk/Wv (WMMA); butterfly q.k dot; online softmax accumulate
    {
      v16h a0 = load_afrag(hT, lane, 0);
      v16h a1 = load_afrag(hT, lane, 1);
#pragma unroll
      for (int nt = 0; nt < 4; ++nt) {
        v8f kacc = {};
        kacc = wmma16(a0, load_bfrag(sWk, lane, nt, 0), kacc);
        kacc = wmma16(a1, load_bfrag(sWk, lane, nt, 1), kacc);
        v8f vacc = {};
        vacc = wmma16(a0, load_bfrag(sWv, lane, nt, 0), vacc);
        vacc = wmma16(a1, load_bfrag(sWv, lane, nt, 1), vacc);
#pragma unroll
        for (int vr = 0; vr < 8; ++vr) {
          float p = kacc[vr] * qreg[nt][vr];
          p += __shfl_xor(p, 1, 32);
          p += __shfl_xor(p, 2, 32);
          p += __shfl_xor(p, 4, 32);
          if (okf[vr] > 0.5f) {
            float mo = mst[nt][vr];
            float mn = fmaxf(mo, p);
            float sc = __expf(mo - mn);
            float w = __expf(p - mn);
            lst[nt][vr] = lst[nt][vr] * sc + w;
            ost[nt][vr] = ost[nt][vr] * sc + w * vacc[vr];
            mst[nt][vr] = mn;
          }
        }
      }
    }
  }

  // cross-wave log-sum-exp merge (aliases dead weight/hT LDS)
  __syncthreads();
  float* mP = (float*)smem;            // [8][16][8]
  float* lP = (float*)(smem + 4096);   // [8][16][8]
  float* oP = (float*)(smem + 8192);   // [8][16][64]
#pragma unroll
  for (int nt = 0; nt < 4; ++nt)
#pragma unroll
    for (int vr = 0; vr < 8; ++vr) {
      int row = vr + rh8, col = cl + 16 * nt;
      oP[(wv * 16 + row) * 64 + col] = ost[nt][vr];
      if ((lane & 7) == 0) {
        int head = col >> 3;
        mP[(wv * 16 + row) * 8 + head] = mst[nt][vr];
        lP[(wv * 16 + row) * 8 + head] = lst[nt][vr];
      }
    }
  __syncthreads();
  if (tid < 128) {
    int row = tid >> 3, head = tid & 7;
    float M = -3.0e38f;
    for (int w = 0; w < 8; ++w) M = fmaxf(M, mP[(w * 16 + row) * 8 + head]);
    float L = 0.f;
    float acc[8];
#pragma unroll
    for (int d = 0; d < 8; ++d) acc[d] = 0.f;
    for (int w = 0; w < 8; ++w) {
      float e = __expf(mP[(w * 16 + row) * 8 + head] - M);
      L += e * lP[(w * 16 + row) * 8 + head];
#pragma unroll
      for (int d = 0; d < 8; ++d)
        acc[d] += e * oP[(w * 16 + row) * 64 + head * 8 + d];
    }
    float inv = 1.f / L;
#pragma unroll
    for (int d = 0; d < 8; ++d)
      gO[(it * 16 + row) * 64 + head * 8 + d] = acc[d] * inv;
  }
}

// ---------------- Kernel C: gated output projection ----------------
__global__ void final_kernel(const float* gO, const float* gC,
                             const float* Wo, const float* ihW, const float* ihb,
                             const float* hhW, const float* hhb,
                             const float* sfW, const float* sfb, float* out) {
  int i = blockIdx.x * blockDim.x + threadIdx.x;
  if (i >= NAG) return;
  float o[64], c[64], inp[64];
  for (int d = 0; d < 64; ++d) { o[d] = gO[i * 64 + d]; c[d] = gC[i * 64 + d]; }
  mv64(inp, Wo, o, nullptr);
  for (int d = 0; d < 64; ++d) {
    float a = ihb[d];
    for (int k = 0; k < 64; ++k) a += ihW[d * 64 + k] * inp[k];
    float h = hhb[d];
    for (int k = 0; k < 64; ++k) h += hhW[d * 64 + k] * c[k];
    float g = 1.f / (1.f + __expf(-(a + h)));
    float sf = sfb[d];
    for (int k = 0; k < 64; ++k) sf += sfW[d * 64 + k] * c[k];
    out[i * 64 + d] = inp[d] + g * (sf - inp[d]);
  }
}

// ---------------- Launch ----------------
// Input order (jax pytree flatten, dicts in sorted-key order):
// 0 positions, 1 bos_mask, 2 padding_mask, 3 t,
// 4 Wk, 5 Wo, 6 Wq, 7 Wv,
// 8 center.l1.W, 9 center.l1.b, 10 center.l2.W, 11 center.l2.b, 12 center.l3.W, 13 center.l3.b,
// 14 center.n1.b, 15 center.n1.g, 16 center.n2.b, 17 center.n2.g, 18 center.n3.b, 19 center.n3.g,
// 20 lin_hh.W, 21 lin_hh.b, 22 lin_ih.W, 23 lin_ih.b, 24 lin_self.W, 25 lin_self.b,
// 26 nbr_aggr.l.W, 27 nbr_aggr.l.b, 28 nbr_aggr.n1.b, 29 nbr_aggr.n1.g, 30 nbr_aggr.n2.b, 31 nbr_aggr.n2.g,
// 32 nbr_dxy.l1.W, 33 nbr_dxy.l1.b, 34 nbr_dxy.l2.W, 35 nbr_dxy.l2.b, 36 nbr_dxy.n1.b, 37 nbr_dxy.n1.g,
// 38 nbr_xy.l1.W, 39 nbr_xy.l1.b, 40 nbr_xy.l2.W, 41 nbr_xy.l2.b, 42 nbr_xy.n1.b, 43 nbr_xy.n1.g,
// 44 norm1.b, 45 norm1.g
extern "C" void kernel_launch(void* const* d_in, const int* in_sizes, int n_in,
                              void* d_out, int out_size, void* d_ws, size_t ws_size,
                              hipStream_t stream) {
  const float* positions = (const float*)d_in[0];
  const int* padding = (const int*)d_in[2];
  const int* tptr = (const int*)d_in[3];
  const float* Wk = (const float*)d_in[4];
  const float* Wo = (const float*)d_in[5];
  const float* Wq = (const float*)d_in[6];
  const float* Wv = (const float*)d_in[7];

  float* ws = (float*)d_ws;
  float* gPosT = ws;                 // 2048
  float* gValid = ws + 2048;         // 1024
  float* gC = ws + 3072;             // 65536
  float* gQ = ws + 68608;            // 65536
  float* gEdxy = ws + 134144;        // 65536
  float* gO = ws + 199680;           // 65536

  prep_kernel<<<4, 256, 0, stream>>>(
      positions, padding, tptr,
      (const float*)d_in[8], (const float*)d_in[9], (const float*)d_in[10], (const float*)d_in[11],
      (const float*)d_in[12], (const float*)d_in[13],
      (const float*)d_in[14], (const float*)d_in[15], (const float*)d_in[16], (const float*)d_in[17],
      (const float*)d_in[18], (const float*)d_in[19], (const float*)d_in[44], (const float*)d_in[45],
      Wq,
      (const float*)d_in[32], (const float*)d_in[33], (const float*)d_in[36], (const float*)d_in[37],
      (const float*)d_in[34], (const float*)d_in[35],
      gPosT, gValid, gC, gQ, gEdxy);

  pair_attn_kernel<<<64, 256, 0, stream>>>(
      (const float*)d_in[38], (const float*)d_in[39], (const float*)d_in[42], (const float*)d_in[43],
      (const float*)d_in[40], (const float*)d_in[41],
      (const float*)d_in[26], (const float*)d_in[27], (const float*)d_in[28], (const float*)d_in[29],
      (const float*)d_in[30], (const float*)d_in[31],
      Wk, Wv, gPosT, gValid, gQ, gEdxy, gO);

  final_kernel<<<4, 256, 0, stream>>>(
      gO, gC, Wo,
      (const float*)d_in[22], (const float*)d_in[23],
      (const float*)d_in[20], (const float*)d_in[21],
      (const float*)d_in[24], (const float*)d_in[25],
      (float*)d_out);
}
